// LinearAttention_83236466196836
// MI455X (gfx1250) — matrix operand order repaired and checked by reading, not compiled, not run on hardware
//
#include <hip/hip_runtime.h>
#include <hip/hip_bf16.h>
#include <cstdint>
#include <cstddef>

// ---------------------------------------------------------------------------
// LinearAttention on MI455X (gfx1250): bf16 WMMA pipeline.
//   k1: qkv = x @ Wqkv^T        (bf16 WMMA, fused elu+1, split to q/k/v [B,H,N,D])
//   k2: kv  = K^T V per head    (bf16 WMMA, TDM tensor_load_to_lds staging, k_sum)
//   k3: out = (Q @ kv) / (Q . k_sum)  -> attn [B,N,C] bf16
//   k4: y   = attn @ Wproj^T + b      (bf16 WMMA, async global->LDS A staging)
// ---------------------------------------------------------------------------

#define DIMC   1024
#define NTOK   16384      // B*N = 4*4096
#define QKV_N  3072
#define NB     4
#define NH     16
#define HD     64
#define SEQ    4096

typedef __attribute__((ext_vector_type(16))) __bf16 v16bf;
typedef __attribute__((ext_vector_type(8)))  float  v8f;
typedef __attribute__((ext_vector_type(4)))  unsigned int u32x4;
typedef __attribute__((ext_vector_type(8)))  int    i32x8;
typedef __attribute__((ext_vector_type(4)))  int    i32x4;

union AFragU { v16bf v; uint32_t u[8]; uint4 q[2]; };
union BFragU { v16bf v; uint32_t u[8]; };

__device__ __forceinline__ uint16_t f32_to_bf16(float f) {
  uint32_t u = __float_as_uint(f);
  uint32_t r = u + 0x7FFFu + ((u >> 16) & 1u);   // round-to-nearest-even
  return (uint16_t)(r >> 16);
}
__device__ __forceinline__ uint32_t pack_bf16x2(float lo, float hi) {
  return (uint32_t)f32_to_bf16(lo) | ((uint32_t)f32_to_bf16(hi) << 16);
}
__device__ __forceinline__ float bf16_to_f32(uint16_t h) {
  return __uint_as_float(((uint32_t)h) << 16);
}
__device__ __forceinline__ v8f wmma_bf16(v16bf a, v16bf b, v8f c) {
  return __builtin_amdgcn_wmma_f32_16x16x32_bf16(
      false, a, false, b, (short)0, c, false, false);
}

// TDM 2-D tile load (groups 2/3 zero => 2-D descriptor).  Arity differs per
// toolchain: ROCm 7.2 (clang-22) = 5 args, upstream clang-23 = 6 args.
__device__ __forceinline__ void tdm_load_2d(u32x4 g0, i32x8 g1) {
  i32x4 gz = {0, 0, 0, 0};
#if __clang_major__ >= 23
  i32x8 gz8 = {0, 0, 0, 0, 0, 0, 0, 0};
  __builtin_amdgcn_tensor_load_to_lds(g0, g1, gz, gz, gz8, 0);
#else
  __builtin_amdgcn_tensor_load_to_lds(g0, g1, gz, gz, 0);
#endif
}

// per-lane async global->LDS 128-bit copy (bf16 data, no conversion needed)
__device__ __forceinline__ void async_copy_b128(const void* gptr, void* lptr) {
  const uint32_t lds_a = (uint32_t)(uintptr_t)lptr;   // low 32b of generic = LDS offset
  const uint64_t ga    = (uint64_t)(uintptr_t)gptr;
  asm volatile("global_load_async_to_lds_b128 %0, %1, off"
               :: "v"(lds_a), "v"(ga) : "memory");
}
__device__ __forceinline__ void async_wait0() {
  asm volatile("s_wait_asynccnt 0x0" ::: "memory");
}

// ---------------------------------------------------------------------------
// Kernel 1: qkv projection.  grid (24, 128), block 256.
// Tile: BM=128, BN=128, KC=32.  Each of 8 waves: 64x32 output (4x2 WMMA tiles).
// Staging converts f32 -> bf16 in-flight, so it must pass through VGPRs.
// ---------------------------------------------------------------------------
__global__ __launch_bounds__(256) void qkv_gemm_kernel(
    const float* __restrict__ x, const float* __restrict__ w,
    uint16_t* __restrict__ qb, uint16_t* __restrict__ kb,
    uint16_t* __restrict__ vb)
{
  __shared__ uint16_t xs[128][40];      // x tile (bf16), +8 pad
  __shared__ uint32_t ws2[16][132];     // W^T tile as K-pairs: ws2[kk/2][o]

  const int tid  = threadIdx.x;
  const int lane = tid & 31;
  const int wid  = tid >> 5;
  const int half = lane >> 4;
  const int ln   = lane & 15;
  const int wrow = wid >> 2;            // 0..1
  const int wcol = wid & 3;             // 0..3
  const int gm0  = blockIdx.y * 128;
  const int gn0  = blockIdx.x * 128;

  v8f acc[4][2];
  for (int i = 0; i < 4; ++i)
    for (int j = 0; j < 2; ++j)
      for (int e = 0; e < 8; ++e) acc[i][j][e] = 0.0f;

  for (int kt = 0; kt < DIMC / 32; ++kt) {
    const int kc0 = kt * 32;
    __syncthreads();
    if (kt + 1 < DIMC / 32)   // global_prefetch_b8 of next x slice
      __builtin_prefetch(x + (size_t)(gm0 + (tid >> 3)) * DIMC + kc0 + 32, 0, 0);
    // stage x tile: 128x32 f32 -> bf16 (1024 float4 / 256 threads)
    for (int i = 0; i < 4; ++i) {
      const int idx = tid + i * 256;
      const int r = idx >> 3, c4 = (idx & 7) << 2;
      const float4 xv = *(const float4*)(x + (size_t)(gm0 + r) * DIMC + kc0 + c4);
      *(uint32_t*)&xs[r][c4]     = pack_bf16x2(xv.x, xv.y);
      *(uint32_t*)&xs[r][c4 + 2] = pack_bf16x2(xv.z, xv.w);
    }
    // stage W tile transposed into K-pair layout (B-fragment friendly)
    for (int i = 0; i < 4; ++i) {
      const int idx = tid + i * 256;
      const int o = idx >> 3, c4 = (idx & 7) << 2;
      const float4 wv = *(const float4*)(w + (size_t)(gn0 + o) * DIMC + kc0 + c4);
      ws2[(c4 >> 1)][o]     = pack_bf16x2(wv.x, wv.y);
      ws2[(c4 >> 1) + 1][o] = pack_bf16x2(wv.z, wv.w);
    }
    __syncthreads();

    AFragU a[4];
    for (int mt = 0; mt < 4; ++mt) {
      const int m = wrow * 64 + mt * 16 + ln;
      a[mt].q[0] = *(const uint4*)&xs[m][half * 8];
      a[mt].q[1] = *(const uint4*)&xs[m][16 + half * 8];
    }
    BFragU bb[2];
    for (int nt = 0; nt < 2; ++nt) {
      const int o = wcol * 32 + nt * 16 + ln;
      for (int j = 0; j < 8; ++j) bb[nt].u[j] = ws2[((j < 4) ? 0 : 8) + half * 4 + (j & 3)][o];
    }
    for (int mt = 0; mt < 4; ++mt)
      for (int nt = 0; nt < 2; ++nt)
        acc[mt][nt] = wmma_bf16(a[mt].v, bb[nt].v, acc[mt][nt]);
  }

  // epilogue: split into q/k/v per head, fuse elu(x)+1 for q,k
  for (int mt = 0; mt < 4; ++mt) {
    for (int nt = 0; nt < 2; ++nt) {
      const int o    = gn0 + wcol * 32 + nt * 16 + ln;
      const int part = o >> 10;           // 0:q 1:k 2:v (uniform per 16-tile)
      const int rem  = o & 1023;
      const int h    = rem >> 6;
      const int d    = rem & 63;
      uint16_t* dst = (part == 0) ? qb : (part == 1) ? kb : vb;
      for (int j = 0; j < 8; ++j) {
        const int gm = gm0 + wrow * 64 + mt * 16 + half * 8 + j;
        const int bi = gm >> 12;          // /4096
        const int n  = gm & 4095;
        float val = acc[mt][nt][j];
        if (part < 2) val = (val > 0.0f) ? (val + 1.0f) : __expf(val);
        dst[((size_t)((bi * NH + h) * SEQ + n)) * HD + d] = f32_to_bf16(val);
      }
    }
  }
}

// ---------------------------------------------------------------------------
// Kernel 2: kv[d,e] = sum_n k[n,d] v[n,e]  (64x64x4096 per (b,h)) + k_sum.
// grid (64), block 256.  Wave w: d-tile = w/2, e-half = (w&1)*32 (2 tiles).
// Tile staging via TDM: wave 0 issues tensor_load_to_lds for the k and v
// [32 x 64] bf16 tiles, waits on TENSORcnt, then the block barriers.
// ---------------------------------------------------------------------------
__global__ __launch_bounds__(256) void kv_ksum_kernel(
    const uint16_t* __restrict__ kb, const uint16_t* __restrict__ vb,
    float* __restrict__ kvb, float* __restrict__ ksum)
{
  __shared__ uint16_t ks[32][64];       // contiguous: TDM writes row-major
  __shared__ uint16_t vs[32][64];

  const int tid  = threadIdx.x;
  const int lane = tid & 31;
  const int wid  = tid >> 5;
  const int half = lane >> 4;
  const int ln   = lane & 15;
  const int bh   = blockIdx.x;
  const size_t base = (size_t)bh * SEQ * HD;

  const int dt = wid >> 1;
  const int eb = (wid & 1) * 32;

  const uint32_t ks_lds = (uint32_t)(uintptr_t)&ks[0][0];
  const uint32_t vs_lds = (uint32_t)(uintptr_t)&vs[0][0];

  // D# group 1 (invariant): data_size=2B, tensor 64 x 4096, tile 64 x 32,
  // dim0 stride 64, dim1 stride 64*4096 (unused for 2-D tile).
  i32x8 g1;
  g1[0] = 0x00010000;              // workgroup_mask=0, data_size=1 (2 bytes)
  g1[1] = (int)(HD << 16);         // tensor_dim0 = 64 (elements)
  g1[2] = (int)(SEQ << 16);        // tensor_dim1 = 4096 (low 16 bits)
  g1[3] = (int)(HD << 16);         // tile_dim0 = 64
  g1[4] = 32;                      // tile_dim1 = 32, tile_dim2 = 0
  g1[5] = HD;                      // tensor_dim0_stride = 64
  g1[6] = 0;                       // stride0 hi16 = 0, stride1 lo16 = 0
  g1[7] = (int)((SEQ * HD) >> 16); // tensor_dim1_stride hi bits

  v8f acc[2];
  for (int t = 0; t < 2; ++t)
    for (int e = 0; e < 8; ++e) acc[t][e] = 0.0f;
  float ksum_part = 0.0f;

  for (int nt = 0; nt < SEQ / 32; ++nt) {
    __syncthreads();                       // previous compute done
    if (wid == 0) {                        // one wave drives the TDM
      const uint64_t gk = (uint64_t)(uintptr_t)(kb + base + (size_t)nt * 32 * HD);
      const uint64_t gv = (uint64_t)(uintptr_t)(vb + base + (size_t)nt * 32 * HD);
      u32x4 g0k, g0v;
      g0k[0] = 1u;                         // count=1, user descriptor
      g0k[1] = ks_lds;                     // lds_addr
      g0k[2] = (uint32_t)gk;               // global_addr[31:0]
      g0k[3] = ((uint32_t)(gk >> 32) & 0x01FFFFFFu) | 0x80000000u;  // type=2
      g0v[0] = 1u;
      g0v[1] = vs_lds;
      g0v[2] = (uint32_t)gv;
      g0v[3] = ((uint32_t)(gv >> 32) & 0x01FFFFFFu) | 0x80000000u;
      tdm_load_2d(g0k, g1);
      tdm_load_2d(g0v, g1);
      __builtin_amdgcn_s_wait_tensorcnt(0);
    }
    __syncthreads();                       // tiles visible to all waves

    if (tid < 64) {   // fused k_sum over this n-chunk
      float s = 0.0f;
      for (int r = 0; r < 32; ++r) s += bf16_to_f32(ks[r][tid]);
      ksum_part += s;
    }

    AFragU a;   // A = K^T: A[m=d][kk=n] gathered transposed from LDS
    {
      const int d = dt * 16 + ln;
      for (int j = 0; j < 8; ++j) {
        const int n0 = ((j < 4) ? 0 : 16) + half * 8 + (j & 3) * 2;
        a.u[j] = (uint32_t)ks[n0][d] | ((uint32_t)ks[n0 + 1][d] << 16);
      }
    }
    for (int t = 0; t < 2; ++t) {
      BFragU b;
      const int e = eb + t * 16 + ln;
      for (int j = 0; j < 8; ++j) {
        const int n0 = ((j < 4) ? 0 : 16) + half * 8 + (j & 3) * 2;
        b.u[j] = (uint32_t)vs[n0][e] | ((uint32_t)vs[n0 + 1][e] << 16);
      }
      acc[t] = wmma_bf16(a.v, b.v, acc[t]);
    }
  }

  if (tid < 64) ksum[bh * HD + tid] = ksum_part;

  for (int t = 0; t < 2; ++t)
    for (int j = 0; j < 8; ++j) {
      const int dd = dt * 16 + half * 8 + j;
      const int e  = eb + t * 16 + ln;
      kvb[(size_t)bh * 4096 + dd * HD + e] = acc[t][j];
    }
}

// ---------------------------------------------------------------------------
// Kernel 3: num = Q @ kv ; out = num / (Q . k_sum + 1e-8) -> attn [B,N,C] bf16.
// grid (32, 64) = (n-strips of 128, b*h), block 256; wave = 16-row strip.
// ---------------------------------------------------------------------------
__global__ __launch_bounds__(256) void attn_out_kernel(
    const uint16_t* __restrict__ qb, const float* __restrict__ kvb,
    const float* __restrict__ ksum, uint16_t* __restrict__ attn)
{
  __shared__ uint32_t kvp[32][68];   // kv as K-pairs: kvp[d/2][e]
  __shared__ float kss[64];
  __shared__ float qks[128];

  const int tid  = threadIdx.x;
  const int lane = tid & 31;
  const int wid  = tid >> 5;
  const int half = lane >> 4;
  const int ln   = lane & 15;
  const int bh   = blockIdx.y;
  const int b    = bh >> 4, h = bh & 15;
  const int n0   = blockIdx.x * 128;
  const size_t qbase = (size_t)bh * SEQ * HD;

  for (int p = tid; p < 32 * 64; p += 256) {
    const int pr = p >> 6, e = p & 63;
    const float lo = kvb[(size_t)bh * 4096 + (2 * pr) * HD + e];
    const float hi = kvb[(size_t)bh * 4096 + (2 * pr + 1) * HD + e];
    kvp[pr][e] = pack_bf16x2(lo, hi);
  }
  if (tid < 64) kss[tid] = ksum[bh * HD + tid];
  __syncthreads();

  if (tid < 128) {   // denominator q . k_sum, one thread per row
    const uint16_t* qrow = qb + qbase + (size_t)(n0 + tid) * HD;
    float s = 0.0f;
    for (int d = 0; d < HD; ++d) s += bf16_to_f32(qrow[d]) * kss[d];
    qks[tid] = s;
  }
  __syncthreads();

  const int m = n0 + wid * 16 + ln;
  const uint16_t* qrow = qb + qbase + (size_t)m * HD;
  AFragU a0, a1;            // K = 64 -> two 16x32 A fragments
  a0.q[0] = *(const uint4*)(qrow + half * 8);
  a0.q[1] = *(const uint4*)(qrow + 16 + half * 8);
  a1.q[0] = *(const uint4*)(qrow + 32 + half * 8);
  a1.q[1] = *(const uint4*)(qrow + 48 + half * 8);

  for (int et = 0; et < 4; ++et) {
    BFragU b0, b1;
    const int e = et * 16 + ln;
    for (int j = 0; j < 8; ++j) {
      b0.u[j] = kvp[((j < 4) ? 0 : 8) + half * 4 + (j & 3)][e];
      b1.u[j] = kvp[16 + ((j < 4) ? 0 : 8) + half * 4 + (j & 3)][e];
    }
    v8f acc;
    for (int q = 0; q < 8; ++q) acc[q] = 0.0f;
    acc = wmma_bf16(a0.v, b0.v, acc);
    acc = wmma_bf16(a1.v, b1.v, acc);

    for (int j = 0; j < 8; ++j) {
      const int row = wid * 16 + half * 8 + j;
      const float val = acc[j] / (qks[row] + 1e-8f);
      attn[(size_t)(b * SEQ + n0 + row) * DIMC + h * HD + e] = f32_to_bf16(val);
    }
  }
}

// ---------------------------------------------------------------------------
// Kernel 4: y = attn @ Wproj^T + b  (f32 out).  grid (8, 128), block 256.
// A tile (already bf16) staged with per-lane async global->LDS b128 copies.
// ---------------------------------------------------------------------------
__global__ __launch_bounds__(256) void proj_gemm_kernel(
    const uint16_t* __restrict__ attn, const float* __restrict__ w,
    const float* __restrict__ bias, float* __restrict__ out)
{
  __shared__ uint16_t xs[128][32];      // contiguous for async b128 stores
  __shared__ uint32_t ws2[16][132];

  const int tid  = threadIdx.x;
  const int lane = tid & 31;
  const int wid  = tid >> 5;
  const int half = lane >> 4;
  const int ln   = lane & 15;
  const int wrow = wid >> 2;
  const int wcol = wid & 3;
  const int gm0  = blockIdx.y * 128;
  const int gn0  = blockIdx.x * 128;

  v8f acc[4][2];
  for (int i = 0; i < 4; ++i)
    for (int j = 0; j < 2; ++j)
      for (int e = 0; e < 8; ++e) acc[i][j][e] = 0.0f;

  for (int kt = 0; kt < DIMC / 32; ++kt) {
    const int kc0 = kt * 32;
    __syncthreads();
    // stage attn tile (bf16): async global -> LDS, 512 x b128 / 256 lanes
    for (int i = 0; i < 2; ++i) {
      const int idx = tid + i * 256;
      const int r = idx >> 2, c8 = (idx & 3) << 3;
      async_copy_b128(attn + (size_t)(gm0 + r) * DIMC + kc0 + c8, &xs[r][c8]);
    }
    // stage W tile transposed + converted into K-pair layout
    for (int i = 0; i < 4; ++i) {
      const int idx = tid + i * 256;
      const int o = idx >> 3, c4 = (idx & 7) << 2;
      const float4 wv = *(const float4*)(w + (size_t)(gn0 + o) * DIMC + kc0 + c4);
      ws2[(c4 >> 1)][o]     = pack_bf16x2(wv.x, wv.y);
      ws2[(c4 >> 1) + 1][o] = pack_bf16x2(wv.z, wv.w);
    }
    async_wait0();                 // ASYNCcnt==0: LDS writes landed
    __syncthreads();

    AFragU a[4];
    for (int mt = 0; mt < 4; ++mt) {
      const int m = wrow * 64 + mt * 16 + ln;
      a[mt].q[0] = *(const uint4*)&xs[m][half * 8];
      a[mt].q[1] = *(const uint4*)&xs[m][16 + half * 8];
    }
    BFragU bb[2];
    for (int nt = 0; nt < 2; ++nt) {
      const int o = wcol * 32 + nt * 16 + ln;
      for (int j = 0; j < 8; ++j) bb[nt].u[j] = ws2[((j < 4) ? 0 : 8) + half * 4 + (j & 3)][o];
    }
    for (int mt = 0; mt < 4; ++mt)
      for (int nt = 0; nt < 2; ++nt)
        acc[mt][nt] = wmma_bf16(a[mt].v, bb[nt].v, acc[mt][nt]);
  }

  for (int mt = 0; mt < 4; ++mt) {
    for (int nt = 0; nt < 2; ++nt) {
      const int o = gn0 + wcol * 32 + nt * 16 + ln;
      const float bv = bias[o];
      for (int j = 0; j < 8; ++j) {
        const int gm = gm0 + wrow * 64 + mt * 16 + half * 8 + j;
        out[(size_t)gm * DIMC + o] = acc[mt][nt][j] + bv;
      }
    }
  }
}

// ---------------------------------------------------------------------------
extern "C" void kernel_launch(void* const* d_in, const int* in_sizes, int n_in,
                              void* d_out, int out_size, void* d_ws, size_t ws_size,
                              hipStream_t stream) {
  const float* x      = (const float*)d_in[0];
  const float* w_qkv  = (const float*)d_in[1];
  const float* w_proj = (const float*)d_in[2];
  const float* b_proj = (const float*)d_in[3];
  float* out = (float*)d_out;

  char* ws = (char*)d_ws;
  const size_t NE = (size_t)NTOK * DIMC;          // 16.7M elements
  uint16_t* qb   = (uint16_t*)(ws + 0 * NE);      // 32 MB bf16 [B,H,N,D]
  uint16_t* kb   = (uint16_t*)(ws + 2 * NE);      // 32 MB
  uint16_t* vb   = (uint16_t*)(ws + 4 * NE);      // 32 MB
  float*    kvb  = (float*)   (ws + 6 * NE);      // 1 MB  [B*H,64,64] f32
  float*    ksum = (float*)   (ws + 6 * NE + (size_t)64 * 4096 * 4);  // 16 KB
  uint16_t* attn = (uint16_t*)(ws + 6 * NE + (size_t)64 * 4096 * 4 + 64 * 64 * 4);

  qkv_gemm_kernel<<<dim3(QKV_N / 128, NTOK / 128), 256, 0, stream>>>(
      x, w_qkv, qb, kb, vb);
  kv_ksum_kernel<<<dim3(NB * NH), 256, 0, stream>>>(kb, vb, kvb, ksum);
  attn_out_kernel<<<dim3(SEQ / 128, NB * NH), 256, 0, stream>>>(
      qb, kvb, ksum, attn);
  proj_gemm_kernel<<<dim3(DIMC / 128, NTOK / 128), 256, 0, stream>>>(
      attn, w_proj, b_proj, out);
}
